// DE_MultiHeadAttention_64647847739755
// MI455X (gfx1250) — matrix operand order repaired and checked by reading, not compiled, not run on hardware
//
#include <hip/hip_runtime.h>
#include <math.h>

// ---------------- types ----------------
typedef __bf16 bf16;
typedef bf16  bf16x8  __attribute__((ext_vector_type(8)));
typedef bf16  bf16x16 __attribute__((ext_vector_type(16)));
typedef float f32x8   __attribute__((ext_vector_type(8)));

static constexpr int Bsz = 8, Seq = 1024, Dm = 1024, Hn = 16, Hd = 64;
static constexpr int SROWF = 1032;  // padded f32 LDS row (4128B: 16B-aligned, +8 banks/row)
static constexpr int SROWP = 1040;  // padded bf16 LDS row (2080B: 16B-aligned, +8 banks/row)
static constexpr int BROW  = 40;    // bf16 elems per LDS B-tile row (80B: conflict-free b128 reads)

// ---------------- helpers ----------------
__device__ __forceinline__ bf16 f2bf(float f) {
  unsigned u = __builtin_bit_cast(unsigned, f);
  u += 0x7FFFu + ((u >> 16) & 1u);              // round-to-nearest-even
  unsigned short h = (unsigned short)(u >> 16);
  return __builtin_bit_cast(bf16, h);
}

// A fragment: 16x32 bf16 row-major, lane(0-15)=M row, 16 contiguous K values at k0 + (lane>=16 ? 16 : 0)
__device__ __forceinline__ bf16x16 load_a_frag(const bf16* __restrict__ base, int ld,
                                               int m0, int k0, int lane) {
  const bf16* p = base + (size_t)(m0 + (lane & 15)) * ld + k0 + ((lane & 16) ? 16 : 0);
  bf16x8 lo = *(const bf16x8*)p;
  bf16x8 hi = *(const bf16x8*)(p + 8);
  return __builtin_shufflevector(lo, hi, 0,1,2,3,4,5,6,7,8,9,10,11,12,13,14,15);
}

// B fragment: 32x16 where B[k][n] = base[n*ld + k] (x@W.T layout):
// lane(0-15)=N col, 16 contiguous K values at k0 + (lane>=16 ? 16 : 0)
__device__ __forceinline__ bf16x16 load_b_frag(const bf16* __restrict__ base, int ld,
                                               int n0, int k0, int lane) {
  const bf16* p = base + (size_t)(n0 + (lane & 15)) * ld + k0 + ((lane & 16) ? 16 : 0);
  bf16x8 lo = *(const bf16x8*)p;
  bf16x8 hi = *(const bf16x8*)(p + 8);
  return __builtin_shufflevector(lo, hi, 0,1,2,3,4,5,6,7,8,9,10,11,12,13,14,15);
}

__device__ __forceinline__ f32x8 wmma_bf16(bf16x16 a, bf16x16 b, f32x8 c) {
  return __builtin_amdgcn_wmma_f32_16x16x32_bf16(false, a, false, b, (short)0, c, false, false);
}

// ---------------- kernel 1: f32 -> bf16 cast ----------------
__global__ void cast_f32_to_bf16(const float* __restrict__ src, bf16* __restrict__ dst, int n) {
  int i = blockIdx.x * blockDim.x + threadIdx.x;
  int stride = gridDim.x * blockDim.x;
  for (; i < n; i += stride) dst[i] = f2bf(src[i]);
}

// ---------------- kernel 2: QKV projection, y = x @ W.T ----------------
// grid(16,64) block(128): block tile M=128 x N=64; each wave owns a 32x64 sub-tile.
// The shared 64x32 B-tile of W is staged in LDS once per block per k-step via
// CDNA5 async global->LDS copies (ASYNCcnt), double-buffered so copy overlaps WMMA.
// transposed==0: dst[b][h][s][d] (Q,K)   transposed==1: dst[b][h][d][s] (V)
__global__ void proj_gemm_kernel(const bf16* __restrict__ X, const bf16* __restrict__ W,
                                 bf16* __restrict__ dst, int transposed) {
  __shared__ bf16 sB[2][64 * BROW];   // 2 x 64 rows x 40 bf16 (80B stride) = 10 KB

  int tid  = threadIdx.x;
  int lane = tid & 31, wv = tid >> 5;
  int m0 = blockIdx.y * 128 + wv * 32;
  int n0 = blockIdx.x * 64;
  unsigned sB_lds = (unsigned)(size_t)(&sB[0][0]);  // LDS byte offset (low 32 bits of aperture)

  // Issue this thread's 2 async B128 copies of W[n0..n0+63][k0..k0+31] into buffer `buf`.
  auto issue_copy = [&](int k0, int buf) {
#pragma unroll
    for (int c = 0; c < 2; ++c) {
      int chunk = tid + c * 128;          // 256 x 16B chunks
      int rowc = chunk >> 2, piece = chunk & 3;
      unsigned long long ga =
          (unsigned long long)(size_t)(W + (size_t)(n0 + rowc) * Dm + k0 + piece * 8);
      unsigned la = sB_lds + (unsigned)(buf * (64 * BROW) + rowc * BROW + piece * 8) * 2u;
      asm volatile("global_load_async_to_lds_b128 %0, %1, off"
                   :: "v"(la), "v"(ga) : "memory");
    }
  };

  issue_copy(0, 0);
  f32x8 acc[2][4] = {};
  for (int k0 = 0; k0 < Dm; k0 += 32) {
    int buf = (k0 >> 5) & 1;
    __syncthreads();                       // everyone done reading buf^1 (prev iteration)
    bool more = (k0 + 32) < Dm;
    if (more) {
      issue_copy(k0 + 32, buf ^ 1);
      asm volatile("s_wait_asynccnt 0x2" ::: "memory");  // current buf's copies landed
    } else {
      asm volatile("s_wait_asynccnt 0x0" ::: "memory");
    }
    __syncthreads();                       // all waves' copies to buf visible

    if (more)
      __builtin_prefetch(X + (size_t)(m0 + (lane & 15)) * Dm + k0 + 32, 0, 1);
    bf16x16 a0 = load_a_frag(X, Dm, m0,      k0, lane);
    bf16x16 a1 = load_a_frag(X, Dm, m0 + 16, k0, lane);
    const bf16* sbuf = &sB[buf][0];
#pragma unroll
    for (int t = 0; t < 4; ++t) {
      // B fragment from LDS tile: row = local n col, 16 contiguous K at (lane>=16 ? 16 : 0)
      const bf16* p = sbuf + (t * 16 + (lane & 15)) * BROW + ((lane & 16) ? 16 : 0);
      bf16x8 blo = *(const bf16x8*)p;
      bf16x8 bhi = *(const bf16x8*)(p + 8);
      bf16x16 bfrag = __builtin_shufflevector(blo, bhi,
          0,1,2,3,4,5,6,7,8,9,10,11,12,13,14,15);
      acc[0][t] = wmma_bf16(a0, bfrag, acc[0][t]);
      acc[1][t] = wmma_bf16(a1, bfrag, acc[1][t]);
    }
  }

#pragma unroll
  for (int mt = 0; mt < 2; ++mt) {
    int rowb = m0 + mt * 16 + ((lane & 16) ? 8 : 0);
#pragma unroll
    for (int t = 0; t < 4; ++t) {
      int col = n0 + t * 16 + (lane & 15);
      int h = col >> 6, d = col & 63;
#pragma unroll
      for (int r = 0; r < 8; ++r) {
        int row = rowb + r;
        int b = row >> 10, s = row & 1023;
        bf16 v = f2bf(acc[mt][t][r]);
        size_t idx = transposed
            ? ((((size_t)b * Hn + h) * Hd + d) * Seq + s)
            : ((((size_t)b * Hn + h) * Seq + s) * Hd + d);
        dst[idx] = v;
      }
    }
  }
}

// ---------------- kernel 3: attention (scores, softmax, attn_dist, P@V) ----------------
// grid(64,128) block(128): blockIdx.x = 16-query tile, blockIdx.y = b*16+h.
// Dynamic LDS: scores f32 16x1032 | P bf16 16x1040 | reduction 16x8 f32.
__global__ void attn_kernel(const bf16* __restrict__ Qp, const bf16* __restrict__ Kp,
                            const bf16* __restrict__ Vt, const int* __restrict__ mask,
                            float* __restrict__ attn_out, bf16* __restrict__ Ctx) {
  extern __shared__ char smem[];
  float* sS  = (float*)smem;                                      // 16 x SROWF f32
  bf16*  sP  = (bf16*)(smem + 16 * SROWF * 4);                    // 16 x SROWP bf16
  float* sRd = (float*)(smem + 16 * SROWF * 4 + 16 * SROWP * 2);  // 16 x 8 f32

  int lane = threadIdx.x & 31, wv = threadIdx.x >> 5;
  int bh = blockIdx.y;
  int b = bh >> 4, h = bh & 15;
  int q0 = blockIdx.x * 16;
  const bf16* Qh = Qp + (size_t)bh * Seq * Hd;  // [s][d]
  const bf16* Kh = Kp + (size_t)bh * Seq * Hd;  // [s][d]
  const bf16* Vh = Vt + (size_t)bh * Hd * Seq;  // [d][s]

  // ---- scores = Q K^T : wave wv owns keys [wv*256, wv*256+256) ----
  bf16x16 aq0 = load_a_frag(Qh, Hd, q0, 0, lane);
  bf16x16 aq1 = load_a_frag(Qh, Hd, q0, 32, lane);
  int rowb = (lane & 16) ? 8 : 0;
  for (int t = 0; t < 16; ++t) {
    int n0 = wv * 256 + t * 16;
    f32x8 cs = {};
    cs = wmma_bf16(aq0, load_b_frag(Kh, Hd, n0, 0, lane), cs);
    cs = wmma_bf16(aq1, load_b_frag(Kh, Hd, n0, 32, lane), cs);
    int col = n0 + (lane & 15);
#pragma unroll
    for (int r = 0; r < 8; ++r) sS[(rowb + r) * SROWF + col] = cs[r];
  }
  __syncthreads();

  // ---- masked scale + softmax: 8 threads/row, interleaved float4 chunks (coalesced) ----
  int row = threadIdx.x >> 3;
  int seg = threadIdx.x & 7;
  float* sSrow = sS + row * SROWF;
  const int4* mrow4 =
      (const int4*)(mask + ((size_t)b * Seq + (q0 + row)) * Seq);
  const float scale = 0.03125f;  // 1/sqrt(D) = 1/32 (reference scales by sqrt(hidden))
  const float NEG = -1.0e9f;
  float mx = -3.0e38f;
  for (int ci = 0; ci < 32; ++ci) {
    int c4 = seg + ci * 8;
    int4  m = mrow4[c4];
    float4* p4 = (float4*)sSrow + c4;
    float4 v = *p4;
    v.x = m.x ? v.x * scale : NEG;
    v.y = m.y ? v.y * scale : NEG;
    v.z = m.z ? v.z * scale : NEG;
    v.w = m.w ? v.w * scale : NEG;
    *p4 = v;
    mx = fmaxf(mx, fmaxf(fmaxf(v.x, v.y), fmaxf(v.z, v.w)));
  }
  sRd[row * 8 + seg] = mx;
  __syncthreads();
  mx = sRd[row * 8];
#pragma unroll
  for (int i = 1; i < 8; ++i) mx = fmaxf(mx, sRd[row * 8 + i]);
  __syncthreads();
  float sum = 0.f;
  for (int ci = 0; ci < 32; ++ci) {
    int c4 = seg + ci * 8;
    float4* p4 = (float4*)sSrow + c4;
    float4 v = *p4;
    v.x = __expf(v.x - mx); v.y = __expf(v.y - mx);
    v.z = __expf(v.z - mx); v.w = __expf(v.w - mx);
    *p4 = v;
    sum += (v.x + v.y) + (v.z + v.w);
  }
  sRd[row * 8 + seg] = sum;
  __syncthreads();
  sum = 0.f;
#pragma unroll
  for (int i = 0; i < 8; ++i) sum += sRd[row * 8 + i];
  float inv = 1.0f / sum;
  float4* arow4 = (float4*)(attn_out + ((size_t)bh * Seq + (q0 + row)) * Seq);
  bf16* sProw = sP + row * SROWP;
  for (int ci = 0; ci < 32; ++ci) {
    int c4 = seg + ci * 8;
    float4 v = *((float4*)sSrow + c4);
    v.x *= inv; v.y *= inv; v.z *= inv; v.w *= inv;
    arow4[c4] = v;                         // coalesced 512MB attn_dist stream (HBM floor)
    int cb = c4 * 4;
    sProw[cb + 0] = f2bf(v.x); sProw[cb + 1] = f2bf(v.y);
    sProw[cb + 2] = f2bf(v.z); sProw[cb + 3] = f2bf(v.w);
  }
  __syncthreads();

  // ---- ctx = P @ V : wave wv owns head-dim tile [wv*16, wv*16+16), K=1024 ----
  int d0 = wv * 16;
  f32x8 cv = {};
  for (int k0 = 0; k0 < Seq; k0 += 32) {
    if (k0 + 32 < Seq)
      __builtin_prefetch(Vh + (size_t)(d0 + (lane & 15)) * Seq + k0 + 32, 0, 1);
    cv = wmma_bf16(load_a_frag(sP, SROWP, 0, k0, lane),
                   load_b_frag(Vh, Seq, d0, k0, lane), cv);
  }
  int col = d0 + (lane & 15);
#pragma unroll
  for (int r = 0; r < 8; ++r) {
    int q = q0 + rowb + r;
    Ctx[((size_t)b * Seq + q) * Dm + h * Hd + col] = f2bf(cv[r]);
  }
}

// ---------------- kernel 4: out = Ctx @ Wo^T + resid, then LayerNorm ----------------
// grid(512) block(128): 16 full rows per workgroup (LN needs whole rows).
// Each wave covers cols [wv*256, wv*256+256) in 4 groups of 4 n-tiles sharing one A frag/k-step.
__global__ void out_ln_kernel(const bf16* __restrict__ Ctx, const bf16* __restrict__ Wo,
                              const float* __restrict__ resid, const float* __restrict__ gamma,
                              const float* __restrict__ beta, float* __restrict__ out) {
  extern __shared__ char smem[];
  float* sO  = (float*)smem;                    // 16 x SROWF f32
  float* sRd = (float*)(smem + 16 * SROWF * 4); // 16 x 8 x 2 f32

  int lane = threadIdx.x & 31, wv = threadIdx.x >> 5;
  int m0 = blockIdx.x * 16;
  int rowb = (lane & 16) ? 8 : 0;
  for (int g = 0; g < 4; ++g) {
    int ng = wv * 256 + g * 64;
    f32x8 acc[4] = {};
    for (int k0 = 0; k0 < Dm; k0 += 32) {
      if (k0 + 32 < Dm)
        __builtin_prefetch(Ctx + (size_t)(m0 + (lane & 15)) * Dm + k0 + 32, 0, 1);
      bf16x16 a = load_a_frag(Ctx, Dm, m0, k0, lane);
#pragma unroll
      for (int t = 0; t < 4; ++t)
        acc[t] = wmma_bf16(a, load_b_frag(Wo, Dm, ng + t * 16, k0, lane), acc[t]);
    }
#pragma unroll
    for (int t = 0; t < 4; ++t) {
      int col = ng + t * 16 + (lane & 15);
#pragma unroll
      for (int r = 0; r < 8; ++r) sO[(rowb + r) * SROWF + col] = acc[t][r];
    }
  }
  __syncthreads();

  // ---- residual + LayerNorm: 8 threads/row, interleaved float4 chunks ----
  int row = threadIdx.x >> 3, seg = threadIdx.x & 7;
  float* sOrow = sO + row * SROWF;
  const float4* rrow4 = (const float4*)(resid + (size_t)(m0 + row) * Dm);
  float s1 = 0.f, s2 = 0.f;
  for (int ci = 0; ci < 32; ++ci) {
    int c4 = seg + ci * 8;
    float4* p4 = (float4*)sOrow + c4;
    float4 v = *p4;
    float4 rr = rrow4[c4];
    v.x += rr.x; v.y += rr.y; v.z += rr.z; v.w += rr.w;
    *p4 = v;
    s1 += (v.x + v.y) + (v.z + v.w);
    s2 += (v.x * v.x + v.y * v.y) + (v.z * v.z + v.w * v.w);
  }
  sRd[row * 8 + seg] = s1;
  sRd[128 + row * 8 + seg] = s2;
  __syncthreads();
  s1 = 0.f; s2 = 0.f;
#pragma unroll
  for (int i = 0; i < 8; ++i) { s1 += sRd[row * 8 + i]; s2 += sRd[128 + row * 8 + i]; }
  float mu  = s1 * (1.0f / 1024.0f);
  float var = s2 * (1.0f / 1024.0f) - mu * mu;
  float inv = rsqrtf(var + 1e-6f);
  float4* orow4 = (float4*)(out + (size_t)(m0 + row) * Dm);
  const float4* g4 = (const float4*)gamma;
  const float4* b4 = (const float4*)beta;
  for (int ci = 0; ci < 32; ++ci) {
    int c4 = seg + ci * 8;
    float4 v = *((float4*)sOrow + c4);
    float4 gg = g4[c4], bb = b4[c4];
    float4 o;
    o.x = (v.x - mu) * inv * gg.x + bb.x;
    o.y = (v.y - mu) * inv * gg.y + bb.y;
    o.z = (v.z - mu) * inv * gg.z + bb.z;
    o.w = (v.w - mu) * inv * gg.w + bb.w;
    orow4[c4] = o;
  }
}

// ---------------- host launcher ----------------
extern "C" void kernel_launch(void* const* d_in, const int* in_sizes, int n_in,
                              void* d_out, int out_size, void* d_ws, size_t ws_size,
                              hipStream_t stream) {
  (void)in_sizes; (void)n_in; (void)out_size; (void)ws_size;
  const float* Qin   = (const float*)d_in[0];
  const float* KVin  = (const float*)d_in[1];
  const int*   mask  = (const int*)d_in[2];
  const float* Wq    = (const float*)d_in[3];
  const float* Wk    = (const float*)d_in[4];
  const float* Wv    = (const float*)d_in[5];
  const float* Wo    = (const float*)d_in[6];
  const float* gamma = (const float*)d_in[7];
  const float* beta  = (const float*)d_in[8];

  float* out      = (float*)d_out;
  float* attn_out = out + (size_t)Bsz * Seq * Dm;   // tuple part 2: (B,H,S,S) f32

  char* ws = (char*)d_ws;
  bf16* Qin_bf  = (bf16*)(ws + 0);                    // 16MB (reused as Ctx later)
  bf16* KVin_bf = (bf16*)(ws + ((size_t)16 << 20));   // 16MB
  bf16* Qp      = (bf16*)(ws + ((size_t)32 << 20));   // 16MB [b][h][s][d]
  bf16* Kp      = (bf16*)(ws + ((size_t)48 << 20));   // 16MB [b][h][s][d]
  bf16* Vt      = (bf16*)(ws + ((size_t)64 << 20));   // 16MB [b][h][d][s]
  bf16* Wq_bf   = (bf16*)(ws + ((size_t)80 << 20));   // 2MB each
  bf16* Wk_bf   = (bf16*)(ws + ((size_t)82 << 20));
  bf16* Wv_bf   = (bf16*)(ws + ((size_t)84 << 20));
  bf16* Wo_bf   = (bf16*)(ws + ((size_t)86 << 20));
  bf16* Ctx     = Qin_bf;  // Qin_bf dead after projections; reuse for context

  const int nTok = Bsz * Seq * Dm;   // 8M
  const int nW   = Dm * Dm;          // 1M
  cast_f32_to_bf16<<<4096, 256, 0, stream>>>(Qin,  Qin_bf,  nTok);
  cast_f32_to_bf16<<<4096, 256, 0, stream>>>(KVin, KVin_bf, nTok);
  cast_f32_to_bf16<<<1024, 256, 0, stream>>>(Wq, Wq_bf, nW);
  cast_f32_to_bf16<<<1024, 256, 0, stream>>>(Wk, Wk_bf, nW);
  cast_f32_to_bf16<<<1024, 256, 0, stream>>>(Wv, Wv_bf, nW);
  cast_f32_to_bf16<<<1024, 256, 0, stream>>>(Wo, Wo_bf, nW);

  const int SMEM_ATTN = 16 * SROWF * 4 + 16 * SROWP * 2 + 16 * 8 * 4;  // 99840 B
  const int SMEM_LN   = 16 * SROWF * 4 + 2 * 16 * 8 * 4;               // 67072 B
  (void)hipFuncSetAttribute((const void*)attn_kernel,
                            hipFuncAttributeMaxDynamicSharedMemorySize, SMEM_ATTN);
  (void)hipFuncSetAttribute((const void*)out_ln_kernel,
                            hipFuncAttributeMaxDynamicSharedMemorySize, SMEM_LN);

  proj_gemm_kernel<<<dim3(16, 64), 128, 0, stream>>>(Qin_bf,  Wq_bf, Qp, 0);
  proj_gemm_kernel<<<dim3(16, 64), 128, 0, stream>>>(KVin_bf, Wk_bf, Kp, 0);
  proj_gemm_kernel<<<dim3(16, 64), 128, 0, stream>>>(KVin_bf, Wv_bf, Vt, 1);

  attn_kernel<<<dim3(64, 128), 128, SMEM_ATTN, stream>>>(Qp, Kp, Vt, mask, attn_out, Ctx);

  out_ln_kernel<<<dim3(512), 128, SMEM_LN, stream>>>(Ctx, Wo_bf, Qin, gamma, beta, out);
}
